// SelfAttention_43628277793268
// MI455X (gfx1250) — matrix operand order repaired and checked by reading, not compiled, not run on hardware
//
#include <hip/hip_runtime.h>

// ---------------- problem constants (from reference) ----------------
constexpr int N_  = 8;
constexpr int C_  = 256;          // channels
constexpr int CH  = 128;          // c = C/2 (qkv inner dim)
constexpr int L_  = 64 * 64;      // tokens = H*W = 4096
constexpr int G_  = 32;           // groupnorm groups
constexpr int CPG = C_ / G_;      // channels per group = 8
constexpr int WSZ = CH * C_;      // one projection weight = 32768 elements

// ---------------- CDNA5 async-copy availability ----------------
#if defined(__has_builtin)
#if __has_builtin(__builtin_amdgcn_global_load_async_to_lds_b128) && \
    __has_builtin(__builtin_amdgcn_s_wait_asynccnt)
#define USE_ASYNC_COPY 1
#endif
#endif
#ifndef USE_ASYNC_COPY
#define USE_ASYNC_COPY 0
#endif

// ---------------- WMMA types (CDNA5 gfx1250, wave32) ----------------
typedef __attribute__((ext_vector_type(16))) __bf16 v16bf;
typedef __attribute__((ext_vector_type(8)))  float  v8f;
typedef __attribute__((ext_vector_type(4)))  int    int4v;

union BF16x16 {           // A/B fragment: 16 bf16 per lane (8 VGPRs)
  v16bf v;
  unsigned short u[16];
  uint4 q2[2];            // two 16B halves (A layout splits K in chunks of 8)
};
union US8 {               // 8 packed bf16 for 16B stores
  unsigned short u[8];
  uint4 q;
};

// native f32 -> bf16 (hardware v_cvt path, RNE)
__device__ __forceinline__ unsigned short f2bf(float f) {
  union { __bf16 b; unsigned short u; } o;
  o.b = (__bf16)f;
  return o.u;
}

__device__ __forceinline__ v8f wmma_bf16(v16bf a, v16bf b, v8f c) {
  // D = A(16x32 bf16) * B(32x16 bf16) + C(16x16 f32)
  return __builtin_amdgcn_wmma_f32_16x16x32_bf16(
      /*neg_a=*/false, a, /*neg_b=*/false, b,
      /*c_mod=*/(short)0, c, /*reuse_a=*/false, /*reuse_b=*/false);
}

#if USE_ASYNC_COPY
// async global -> LDS b128 copy; signature: (AS1 int4*, AS3 int4*, imm, imm)
__device__ __forceinline__ void async_copy_b128(const uint4* gsrc, uint4* ldst) {
  __builtin_amdgcn_global_load_async_to_lds_b128(
      (__attribute__((address_space(1))) int4v*)(gsrc),
      (__attribute__((address_space(3))) int4v*)(ldst), 0, 0);
}
#endif

// ==================================================================
// Kernel 0: one-shot fp32 -> bf16 conversion of all projection weights.
// dst layout: [wq | wk | wv | wo], each 32768 elements.
// ==================================================================
__global__ void wconv_kernel(const float* __restrict__ wq,
                             const float* __restrict__ wk,
                             const float* __restrict__ wv,
                             const float* __restrict__ wo,
                             unsigned short* __restrict__ dst) {
  const int tid = blockIdx.x * blockDim.x + threadIdx.x;   // 0 .. 4*WSZ-1
  const int sel = tid >> 15;
  const int off = tid & (WSZ - 1);
  const float* src = (sel == 0) ? wq : (sel == 1) ? wk : (sel == 2) ? wv : wo;
  dst[tid] = f2bf(src[off]);
}

// ==================================================================
// Kernel 1: GroupNorm statistics. One block per (n, g); the group's
// 8 channels x 4096 tokens are contiguous in x.
// ==================================================================
__global__ void gn_stats_kernel(const float* __restrict__ x,
                                float* __restrict__ stats) {
  const int bid = blockIdx.x;                 // n*G + g
  const float* p = x + (size_t)bid * (CPG * L_);
  float s = 0.f, ss = 0.f;
  for (int i = threadIdx.x; i < CPG * L_; i += blockDim.x) {
    float v = p[i];
    s += v; ss += v * v;
  }
  __shared__ float red0[256];
  __shared__ float red1[256];
  red0[threadIdx.x] = s; red1[threadIdx.x] = ss;
  __syncthreads();
#pragma unroll
  for (int off = 128; off > 0; off >>= 1) {
    if ((int)threadIdx.x < off) {
      red0[threadIdx.x] += red0[threadIdx.x + off];
      red1[threadIdx.x] += red1[threadIdx.x + off];
    }
    __syncthreads();
  }
  if (threadIdx.x == 0) {
    const float inv = 1.0f / (float)(CPG * L_);
    float mean = red0[0] * inv;
    float var  = red1[0] * inv - mean * mean;
    stats[2 * bid]     = mean;
    stats[2 * bid + 1] = rsqrtf(var + 1e-5f);
  }
}

// ==================================================================
// Kernel 2: GN apply + SiLU -> f (bf16), TOKEN-major [N][L][C] so that
// GEMM B-fragments read 16 contiguous channels per lane.
// ==================================================================
__global__ void gn_apply_kernel(const float* __restrict__ x,
                                const float* __restrict__ w,
                                const float* __restrict__ b,
                                const float* __restrict__ stats,
                                unsigned short* __restrict__ fbf) {
  const int ch = blockIdx.x;
  const int n  = blockIdx.y;
  const int g  = ch / CPG;
  const float mean = stats[2 * (n * G_ + g)];
  const float rstd = stats[2 * (n * G_ + g) + 1];
  const float sw = w[ch] * rstd;
  const float sb = b[ch] - mean * sw;
  const float* px = x + ((size_t)n * C_ + ch) * L_;
  unsigned short* pf = fbf + (size_t)n * L_ * C_ + ch;
  for (int t = threadIdx.x; t < L_; t += blockDim.x) {
    float z = px[t] * sw + sb;
    float sil = z / (1.f + __expf(-z));        // SiLU
    pf[(size_t)t * C_] = f2bf(sil);
  }
}

// ==================================================================
// Kernel 3: fused Q/K/V projection GEMM (bf16 WMMA, fp32 accumulate).
// grid = (L/64, CH/16, N*3); block = 128 (4 waves, 1 token-tile each).
// q,k stored token-major [N][L][CH]; v channel-major [N][CH][L].
// Weights come pre-converted to bf16, so A fragments are pure loads.
// ==================================================================
__global__ void qkv_gemm_kernel(const unsigned short* __restrict__ fbf,
                                const unsigned short* __restrict__ wbf,
                                const float* __restrict__ bq,
                                const float* __restrict__ bk,
                                const float* __restrict__ bv,
                                unsigned short* __restrict__ q,
                                unsigned short* __restrict__ k,
                                unsigned short* __restrict__ v) {
  const int lane = threadIdx.x & 31;
  const int wave = threadIdx.x >> 5;
  const int h    = lane >> 4;                  // half-wave select
  const int tt   = blockIdx.x * 4 + wave;      // token tile (16 tokens)
  const int mb   = blockIdx.y;                 // out-channel block (16)
  const int which = blockIdx.z % 3;            // 0:q 1:k 2:v
  const int n     = blockIdx.z / 3;
  const unsigned short* W = wbf + (size_t)which * WSZ;
  const float* B = (which == 0) ? bq : (which == 1) ? bk : bv;

  // A fragments: weight rows, K = C_ = 256 -> 8 chunks of 32.
  // A layout: half-wave h holds 16 contiguous K in {h*16..h*16+15}.
  const int mrow = mb * 16 + (lane & 15);
  const unsigned short* wr = W + (size_t)mrow * C_;
  BF16x16 afrag[8];
#pragma unroll
  for (int kb = 0; kb < 8; ++kb) {
    afrag[kb].q2[0] = *(const uint4*)(wr + kb * 32 + h * 16);
    afrag[kb].q2[1] = *(const uint4*)(wr + kb * 32 + h * 16 + 8);
  }

  // B fragments from f (token-major): lane = token column, half-wave h
  // holds 16 contiguous K (channels) starting at kb*32 + h*16.
  const int tok = tt * 16 + (lane & 15);
  const unsigned short* fb =
      fbf + (size_t)n * L_ * C_ + (size_t)tok * C_ + h * 16;
  v8f acc = {};
#pragma unroll
  for (int kb = 0; kb < 8; ++kb) {
    v16bf bfrag = *(const v16bf*)(fb + kb * 32);
    acc = wmma_bf16(afrag[kb].v, bfrag, acc);
  }

  // D layout: lane = column(token), VGPR r = row m0+r.
  const int m0 = mb * 16 + h * 8;
  if (which < 2) {
    unsigned short* out =
        ((which == 0) ? q : k) + (size_t)n * L_ * CH + (size_t)tok * CH + m0;
    US8 o;
#pragma unroll
    for (int r = 0; r < 8; ++r) o.u[r] = f2bf(acc[r] + B[m0 + r]);
    *(uint4*)out = o.q;
  } else {
    unsigned short* out = v + (size_t)n * CH * L_;
#pragma unroll
    for (int r = 0; r < 8; ++r)
      out[(size_t)(m0 + r) * L_ + tok] = f2bf(acc[r] + B[m0 + r]);
  }
}

// ==================================================================
// Kernel 4: flash attention + fused SiLU.
// grid = (L/64, N); block = 128 (4 waves). Each wave: 16 queries x
// full c=128 fp32 accumulator. 32-key steps staged through LDS via
// CDNA5 async global->LDS copies (ASYNCcnt) when available.
// ==================================================================
__global__ void flash_attn_kernel(const unsigned short* __restrict__ q,
                                  const unsigned short* __restrict__ k,
                                  const unsigned short* __restrict__ v,
                                  unsigned short* __restrict__ f2) {
  __shared__ __align__(64) unsigned short ldsK[32 * 128];   // [key][ch]
  __shared__ __align__(64) unsigned short ldsV[128 * 32];   // [ch][key]
  __shared__ __align__(64) unsigned short ldsP[4][16 * 32]; // per-wave P [row][key]

  const int tid  = threadIdx.x;
  const int lane = tid & 31;
  const int wave = tid >> 5;
  const int h    = lane >> 4;
  const int n    = blockIdx.y;
  const int q0   = (blockIdx.x * 4 + wave) * 16;

  const unsigned short* qn = q + (size_t)n * L_ * CH;
  const unsigned short* kn = k + (size_t)n * L_ * CH;
  const unsigned short* vn = v + (size_t)n * CH * L_;

  // Q fragments: 4 K-chunks of 32 channels (A layout, 16 contiguous per half).
  const unsigned short* qrow = qn + (size_t)(q0 + (lane & 15)) * CH;
  BF16x16 qf[4];
#pragma unroll
  for (int cb = 0; cb < 4; ++cb) {
    qf[cb].q2[0] = *(const uint4*)(qrow + cb * 32 + h * 16);
    qf[cb].q2[1] = *(const uint4*)(qrow + cb * 32 + h * 16 + 8);
  }

  v8f O[8];
#pragma unroll
  for (int i = 0; i < 8; ++i) O[i] = (v8f){};
  float mrow[8], lrow[8];
#pragma unroll
  for (int r = 0; r < 8; ++r) { mrow[r] = -1e30f; lrow[r] = 0.f; }

  for (int j0 = 0; j0 < L_; j0 += 32) {
    __syncthreads();   // previous iteration's LDS reads complete
    // ---- stage K tile (32 keys x 128 ch) + V tile (128 ch x 32 keys) ----
    {
      const uint4* src =
          (const uint4*)(kn + (size_t)(j0 + (tid >> 2)) * CH + (tid & 3) * 32);
      uint4* dst = (uint4*)(ldsK + (tid >> 2) * 128 + (tid & 3) * 32);
      const uint4* vs = (const uint4*)(vn + (size_t)tid * L_ + j0);
      uint4* vd = (uint4*)(ldsV + tid * 32);
#if USE_ASYNC_COPY
#pragma unroll
      for (int i = 0; i < 4; ++i) async_copy_b128(src + i, dst + i);
#pragma unroll
      for (int i = 0; i < 4; ++i) async_copy_b128(vs + i, vd + i);
      __builtin_amdgcn_s_wait_asynccnt(0);   // my async copies landed in LDS
#else
      dst[0] = src[0]; dst[1] = src[1]; dst[2] = src[2]; dst[3] = src[3];
      vd[0] = vs[0]; vd[1] = vs[1]; vd[2] = vs[2]; vd[3] = vs[3];
#endif
      if (j0 + 32 < L_) {   // prefetch next tiles (global_prefetch_b8)
        __builtin_prefetch(kn + (size_t)(j0 + 32 + (tid >> 2)) * CH + (tid & 3) * 32, 0, 1);
        __builtin_prefetch(vn + (size_t)tid * L_ + j0 + 32, 0, 1);
      }
    }
    __syncthreads();   // tiles visible to all waves

    // ---- S = Q * K^T for two 16-key tiles; K-dim = 128 channels ----
    v8f s0 = {}, s1 = {};
#pragma unroll
    for (int cb = 0; cb < 4; ++cb) {
      // B layout: lane = key column, half-wave h holds 16 contiguous channels
      v16bf k0 = *(const v16bf*)(ldsK + (lane & 15) * 128 + cb * 32 + h * 16);
      v16bf k1 = *(const v16bf*)(ldsK + (16 + (lane & 15)) * 128 + cb * 32 + h * 16);
      s0 = wmma_bf16(qf[cb].v, k0, s0);
      s1 = wmma_bf16(qf[cb].v, k1, s1);
    }

    // ---- online softmax (rows live across 16-lane half-wave groups) ----
    float p0[8], p1[8], scale[8];
#pragma unroll
    for (int r = 0; r < 8; ++r) {
      float t = fmaxf(s0[r], s1[r]);
      t = fmaxf(t, __shfl_xor(t, 1, 32));
      t = fmaxf(t, __shfl_xor(t, 2, 32));
      t = fmaxf(t, __shfl_xor(t, 4, 32));
      t = fmaxf(t, __shfl_xor(t, 8, 32));
      float mn = fmaxf(mrow[r], t);
      scale[r] = __expf(mrow[r] - mn);
      mrow[r] = mn;
      p0[r] = __expf(s0[r] - mn);
      p1[r] = __expf(s1[r] - mn);
      float su = p0[r] + p1[r];
      su += __shfl_xor(su, 1, 32);
      su += __shfl_xor(su, 2, 32);
      su += __shfl_xor(su, 4, 32);
      su += __shfl_xor(su, 8, 32);
      lrow[r] = lrow[r] * scale[r] + su;
    }
#pragma unroll
    for (int cb = 0; cb < 8; ++cb)
#pragma unroll
      for (int r = 0; r < 8; ++r) O[cb][r] *= scale[r];

    // ---- transpose P from D-layout to A-layout via wave-private LDS ----
    unsigned short* pw = ldsP[wave];
    const int rbase = h * 8;
#pragma unroll
    for (int r = 0; r < 8; ++r) {
      pw[(rbase + r) * 32 + (lane & 15)]      = f2bf(p0[r]);
      pw[(rbase + r) * 32 + 16 + (lane & 15)] = f2bf(p1[r]);
    }
    // same-wave LDS ops stay in order; compiler inserts s_wait_dscnt
    BF16x16 pf;
    pf.q2[0] = *(const uint4*)(pw + (lane & 15) * 32 + h * 16);
    pf.q2[1] = *(const uint4*)(pw + (lane & 15) * 32 + h * 16 + 8);

    // ---- O += P(16x32) * V(32x128): 8 output tiles ----
#pragma unroll
    for (int cb = 0; cb < 8; ++cb) {
      v16bf vf = *(const v16bf*)(ldsV + (cb * 16 + (lane & 15)) * 32 + h * 16);
      O[cb] = wmma_bf16(pf.v, vf, O[cb]);
    }
  }

  // ---- epilogue: 1/l normalize + SiLU, bf16 token-major [L][CH] ----
  unsigned short* f2n = f2 + (size_t)n * L_ * CH;
  const int rbase = h * 8;
#pragma unroll
  for (int r = 0; r < 8; ++r) {
    const float inv = 1.f / lrow[r];
    const int row = q0 + rbase + r;
#pragma unroll
    for (int cb = 0; cb < 8; ++cb) {
      float val = O[cb][r] * inv;
      float sil = val / (1.f + __expf(-val));
      f2n[(size_t)row * CH + cb * 16 + (lane & 15)] = f2bf(sil);
    }
  }
}

// ==================================================================
// Kernel 5: output projection (C x c GEMM) + bias + residual, fp32 out.
// grid = (L/64, C/16, N); block = 128 (4 waves).
// ==================================================================
__global__ void out_proj_kernel(const unsigned short* __restrict__ f2,
                                const unsigned short* __restrict__ wobf,
                                const float* __restrict__ bo,
                                const float* __restrict__ x,
                                float* __restrict__ out) {
  const int lane = threadIdx.x & 31;
  const int wave = threadIdx.x >> 5;
  const int h    = lane >> 4;
  const int tt   = blockIdx.x * 4 + wave;
  const int mb   = blockIdx.y;                 // 0..15
  const int n    = blockIdx.z;

  const int mrow = mb * 16 + (lane & 15);
  const unsigned short* wr = wobf + (size_t)mrow * CH;
  BF16x16 afrag[4];                            // K = CH = 128 -> 4 chunks
#pragma unroll
  for (int kb = 0; kb < 4; ++kb) {
    afrag[kb].q2[0] = *(const uint4*)(wr + kb * 32 + h * 16);
    afrag[kb].q2[1] = *(const uint4*)(wr + kb * 32 + h * 16 + 8);
  }

  const int tok = tt * 16 + (lane & 15);
  const unsigned short* fb =
      f2 + (size_t)n * L_ * CH + (size_t)tok * CH + h * 16;
  v8f acc = {};
#pragma unroll
  for (int kb = 0; kb < 4; ++kb) {
    v16bf bfrag = *(const v16bf*)(fb + kb * 32);
    acc = wmma_bf16(afrag[kb].v, bfrag, acc);
  }

  const int m0 = mb * 16 + h * 8;
  const float* xn = x + (size_t)n * C_ * L_;
  float* on = out + (size_t)n * C_ * L_;
#pragma unroll
  for (int r = 0; r < 8; ++r) {
    size_t idx = (size_t)(m0 + r) * L_ + tok;
    on[idx] = acc[r] + bo[m0 + r] + xn[idx];
  }
}

// ==================================================================
// Launch
// ==================================================================
extern "C" void kernel_launch(void* const* d_in, const int* in_sizes, int n_in,
                              void* d_out, int out_size, void* d_ws, size_t ws_size,
                              hipStream_t stream) {
  (void)in_sizes; (void)n_in; (void)out_size; (void)ws_size;
  const float* x    = (const float*)d_in[0];
  const float* gn_w = (const float*)d_in[1];
  const float* gn_b = (const float*)d_in[2];
  const float* wq   = (const float*)d_in[3];
  const float* bq   = (const float*)d_in[4];
  const float* wk   = (const float*)d_in[5];
  const float* bk   = (const float*)d_in[6];
  const float* wv   = (const float*)d_in[7];
  const float* bv   = (const float*)d_in[8];
  const float* wo   = (const float*)d_in[9];
  const float* bo   = (const float*)d_in[10];
  float* out = (float*)d_out;

  // workspace layout (bf16 stored as raw ushort); ~50.6 MB total
  unsigned short* fbf = (unsigned short*)d_ws;                 // [N][L][C]
  unsigned short* qb  = fbf + (size_t)N_ * L_ * C_;            // [N][L][CH]
  unsigned short* kb  = qb  + (size_t)N_ * L_ * CH;            // [N][L][CH]
  unsigned short* vb  = kb  + (size_t)N_ * L_ * CH;            // [N][CH][L]
  unsigned short* f2  = vb  + (size_t)N_ * CH * L_;            // [N][L][CH]
  unsigned short* wbf = f2  + (size_t)N_ * L_ * CH;            // [4][32768] bf16 weights
  float* stats = (float*)(wbf + 4 * (size_t)WSZ);              // [N*G][2]

  wconv_kernel<<<dim3(4 * WSZ / 256), dim3(256), 0, stream>>>(wq, wk, wv, wo, wbf);
  gn_stats_kernel<<<dim3(N_ * G_), dim3(256), 0, stream>>>(x, stats);
  gn_apply_kernel<<<dim3(C_, N_), dim3(256), 0, stream>>>(x, gn_w, gn_b, stats, fbf);
  qkv_gemm_kernel<<<dim3(L_ / 64, CH / 16, N_ * 3), dim3(128), 0, stream>>>(
      fbf, wbf, bq, bk, bv, qb, kb, vb);
  flash_attn_kernel<<<dim3(L_ / 64, N_), dim3(128), 0, stream>>>(qb, kb, vb, f2);
  out_proj_kernel<<<dim3(L_ / 64, C_ / 16, N_), dim3(128), 0, stream>>>(
      f2, wbf + 3 * (size_t)WSZ, bo, x, out);
}
